// MGDPR_18107582120672
// MI455X (gfx1250) — matrix operand order repaired and checked by reading, not compiled, not run on hardware
//
#include <hip/hip_runtime.h>

#define L_ 2
#define R_ 4
#define N_ 2048
#define F_ 64
#define TD_ 16
#define S_ 3
#define INTER_ 256
#define OUT_ 2048
#define INDIM_ 32768

#define FSPLIT_ 8                    // K-splits for diffusion GEMM (2048/8 = 256 K per block)
#define FKRANGE_ (N_ / FSPLIT_)
#define KSPL_ 128                    // K-splits for QKV GEMMs (32768/128 = 256 K per block)
#define KRANGE_ (INDIM_ / KSPL_)

typedef float v8f  __attribute__((ext_vector_type(8)));
typedef _Float16 v16h __attribute__((ext_vector_type(16)));
typedef _Float16 v8h  __attribute__((ext_vector_type(8)));

__device__ __forceinline__ float prelu(float x, float a) { return x >= 0.f ? x : a * x; }

__device__ __forceinline__ void ld4(const float* __restrict__ p, float* d) {
    const float4 v = *(const float4*)p;
    d[0] = v.x; d[1] = v.y; d[2] = v.z; d[3] = v.w;
}

// ---------------------------------------------------------------------------
// Kernel 1: featpart[ks] = ((sum_s theta_s * T_s) .* a)[:, kslice] @ h[kslice, :]
// grid (N/64, R, FSPLIT_), block 128 (4 waves). Wave: 16-row tile x 64 cols.
// ---------------------------------------------------------------------------
__global__ __launch_bounds__(128)
void feat_wmma_kernel(const float* __restrict__ h,      // [R,N,F]
                      const float* __restrict__ a,      // [R,N,N]
                      const float* __restrict__ T,      // [R,S,N,N]  (layer slice)
                      const float* __restrict__ theta,  // [R,S]      (layer slice)
                      float* __restrict__ featpart)     // [FSPLIT_][R,N,F]
{
    __shared__ __align__(16) _Float16 sB[F_ * 32];      // transposed: [f][kk]
    const int r     = blockIdx.y;
    const int tid   = threadIdx.x;
    const int wave  = tid >> 5;
    const int lane  = tid & 31;
    const int khalf = lane >> 4;
    const int l15   = lane & 15;
    const int rowbase = blockIdx.x * 64 + wave * 16;
    const int row     = rowbase + l15;
    const int kstart  = blockIdx.z * FKRANGE_;

    const float th0 = theta[r * S_ + 0];
    const float th1 = theta[r * S_ + 1];
    const float th2 = theta[r * S_ + 2];
    const float* T0p = T + ((size_t)(r * S_ + 0) * N_ + row) * N_;
    const float* T1p = T + ((size_t)(r * S_ + 1) * N_ + row) * N_;
    const float* T2p = T + ((size_t)(r * S_ + 2) * N_ + row) * N_;
    const float* ap  = a + ((size_t)r * N_ + row) * N_;
    const float* hr  = h + (size_t)r * N_ * F_;

    v8f acc[4] = {};

    for (int kb = kstart; kb < kstart + FKRANGE_; kb += 32) {
        // prefetch next chunk of the four global streams across the barrier
        if (kb + 32 < kstart + FKRANGE_) {
            const int kn = kb + 32 + khalf * 8;
            __builtin_prefetch(T0p + kn, 0, 3);
            __builtin_prefetch(T1p + kn, 0, 3);
            __builtin_prefetch(T2p + kn, 0, 3);
            __builtin_prefetch(ap  + kn, 0, 3);
        }
        // cooperative load of h chunk [kb:kb+32, 0:64] (contiguous) -> LDS transposed f16
        {
            const float* src = hr + (size_t)kb * F_;
#pragma unroll
            for (int i = 0; i < 4; ++i) {
                const int idx = tid * 16 + i * 4;        // 0..2047
                const float4 v = *(const float4*)(src + idx);
                const int kk = idx >> 6;                 // local k
                const int f  = idx & 63;
                sB[(f + 0) * 32 + kk] = (_Float16)v.x;
                sB[(f + 1) * 32 + kk] = (_Float16)v.y;
                sB[(f + 2) * 32 + kk] = (_Float16)v.z;
                sB[(f + 3) * 32 + kk] = (_Float16)v.w;
            }
        }
        __syncthreads();

        // Build A fragment: fused M = (th0*T0+th1*T1+th2*T2)*a, converted to f16
        v16h afr;
#pragma unroll
        for (int run = 0; run < 2; ++run) {
            const int kk = kb + khalf * 8 + run * 16;
            float t0[8], t1[8], t2[8], av[8];
            ld4(T0p + kk, t0); ld4(T0p + kk + 4, t0 + 4);
            ld4(T1p + kk, t1); ld4(T1p + kk + 4, t1 + 4);
            ld4(T2p + kk, t2); ld4(T2p + kk + 4, t2 + 4);
            ld4(ap  + kk, av); ld4(ap  + kk + 4, av + 4);
#pragma unroll
            for (int j = 0; j < 8; ++j)
                afr[run * 8 + j] = (_Float16)((th0 * t0[j] + th1 * t1[j] + th2 * t2[j]) * av[j]);
        }

        // 4 column tiles of 16
#pragma unroll
        for (int ct = 0; ct < 4; ++ct) {
            const _Float16* bp = sB + (ct * 16 + l15) * 32 + khalf * 8;
            v8h b0 = *(const v8h*)(bp);
            v8h b1 = *(const v8h*)(bp + 16);
            v16h bfr = __builtin_shufflevector(b0, b1, 0,1,2,3,4,5,6,7,8,9,10,11,12,13,14,15);
            acc[ct] = __builtin_amdgcn_wmma_f32_16x16x32_f16(
                false, afr, false, bfr, (short)0, acc[ct], false, false);
        }
        __syncthreads();
    }

    float* out = featpart + (size_t)blockIdx.z * (R_ * N_ * F_);
    const int rofs = khalf * 8;
#pragma unroll
    for (int ct = 0; ct < 4; ++ct)
#pragma unroll
        for (int j = 0; j < 8; ++j)
            out[((size_t)r * N_ + (rowbase + rofs + j)) * F_ + ct * 16 + l15] = acc[ct][j];
}

// ---------------------------------------------------------------------------
// Kernel 2: feat = sum(featpart); diff = prelu(feat @ fc_w.T + fc_b);
//           h_next = prelu(conv_w @ diff + conv_b)
// grid N, block 64 (one n per block, thread = output feature o)
// ---------------------------------------------------------------------------
__global__ __launch_bounds__(64)
void fc_conv_kernel(const float* __restrict__ featpart,   // [FSPLIT_][R,N,F]
                    const float* __restrict__ fc_w, const float* __restrict__ fc_b,
                    const float* __restrict__ conv_w, const float* __restrict__ conv_b,
                    const float* __restrict__ p0p, const float* __restrict__ p1p,
                    float* __restrict__ diff, float* __restrict__ hnext, int write_h)
{
    __shared__ float sf[R_ * F_];
    const int n = blockIdx.x;
    const int o = threadIdx.x;
    {
        const int idx = o * 4;
        const int r = idx >> 6, f = idx & 63;
        const size_t off = ((size_t)r * N_ + n) * F_ + f;
        float sx = 0.f, sy = 0.f, sz = 0.f, sw = 0.f;
#pragma unroll
        for (int p = 0; p < FSPLIT_; ++p) {
            const float4 v = *(const float4*)(featpart + (size_t)p * (R_ * N_ * F_) + off);
            sx += v.x; sy += v.y; sz += v.z; sw += v.w;
        }
        sf[idx] = sx; sf[idx + 1] = sy; sf[idx + 2] = sz; sf[idx + 3] = sw;
    }
    __syncthreads();
    const float p0 = p0p[0], p1 = p1p[0];
    float d[R_];
#pragma unroll
    for (int r = 0; r < R_; ++r) {
        const float* w = fc_w + ((size_t)r * F_ + o) * F_;
        float s = fc_b[r * F_ + o];
#pragma unroll 8
        for (int f = 0; f < F_; ++f) s += sf[r * F_ + f] * w[f];
        d[r] = prelu(s, p0);
        diff[((size_t)r * N_ + n) * F_ + o] = d[r];
    }
    if (write_h) {
#pragma unroll
        for (int q = 0; q < R_; ++q) {
            float s = conv_b[q];
#pragma unroll
            for (int r = 0; r < R_; ++r) s += conv_w[q * R_ + r] * d[r];
            hnext[((size_t)q * N_ + n) * F_ + o] = prelu(s, p1);
        }
    }
}

// ---------------------------------------------------------------------------
// Kernel 3: QKV partial GEMMs (q/k/v merged): part[m][split] = xr[:, ks] @ Wm[:, ks].T
// grid (KSPL_, 16 col-tiles, 3), block 32 (one wave). WMMA f32<-f16.
// ---------------------------------------------------------------------------
__global__ __launch_bounds__(32)
void qkv_wmma_kernel(const float* __restrict__ xr,   // [16, 32768] (diff flat)
                     const float* __restrict__ Wq, const float* __restrict__ Wk,
                     const float* __restrict__ Wv,
                     float* __restrict__ part)       // [3][KSPL_][16*256]
{
    const int m = blockIdx.z;
    const float* W = (m == 0) ? Wq : (m == 1) ? Wk : Wv;

    const int lane  = threadIdx.x;
    const int khalf = lane >> 4;
    const int l15   = lane & 15;
    const int colbase = blockIdx.y * 16;
    const int kstart  = blockIdx.x * KRANGE_;
    const float* xrow = xr + (size_t)l15 * INDIM_;
    const float* wrow = W + (size_t)(colbase + l15) * INDIM_;

    v8f acc = {};
    for (int kb = kstart; kb < kstart + KRANGE_; kb += 32) {
        v16h afr, bfr;
        float ta[8];
#pragma unroll
        for (int run = 0; run < 2; ++run) {
            const float* p = xrow + kb + khalf * 8 + run * 16;
            ld4(p, ta); ld4(p + 4, ta + 4);
#pragma unroll
            for (int j = 0; j < 8; ++j) afr[run * 8 + j] = (_Float16)ta[j];
        }
        float tb[16];
        const float* q = wrow + kb + khalf * 8;
        ld4(q, tb); ld4(q + 4, tb + 4); ld4(q + 16, tb + 8); ld4(q + 20, tb + 12);
#pragma unroll
        for (int e = 0; e < 16; ++e) bfr[e] = (_Float16)tb[e];
        acc = __builtin_amdgcn_wmma_f32_16x16x32_f16(
            false, afr, false, bfr, (short)0, acc, false, false);
    }
    float* o = part + ((size_t)m * KSPL_ + blockIdx.x) * (TD_ * INTER_);
    const int rofs = khalf * 8;
#pragma unroll
    for (int j = 0; j < 8; ++j)
        o[(rofs + j) * INTER_ + colbase + l15] = acc[j];
}

// grid (16, 3), block 256
__global__ __launch_bounds__(256)
void qkv_reduce_kernel(const float* __restrict__ part, float* __restrict__ out)
{
    const int m   = blockIdx.y;
    const int idx = blockIdx.x * blockDim.x + threadIdx.x;    // 0..4095
    const float* p = part + (size_t)m * KSPL_ * (TD_ * INTER_);
    float s = 0.f;
    for (int k = 0; k < KSPL_; ++k) s += p[(size_t)k * (TD_ * INTER_) + idx];
    out[(size_t)m * (TD_ * INTER_) + idx] = s;
}

// ---------------------------------------------------------------------------
// Kernel 4: retention core: ret = (D_gamma .* (q k^T)) @ v  (16x256, one block)
// ---------------------------------------------------------------------------
__global__ __launch_bounds__(256)
void retention_kernel(const float* __restrict__ qb, const float* __restrict__ kbuf,
                      const float* __restrict__ vbuf,
                      const float* __restrict__ bq, const float* __restrict__ bk,
                      const float* __restrict__ bv, float* __restrict__ ret)
{
    __shared__ float sq[TD_ * INTER_];
    __shared__ float sk[TD_ * INTER_];
    __shared__ float ss[TD_ * TD_];
    const int tid = threadIdx.x;
    for (int i = tid; i < TD_ * INTER_; i += 256) {
        const int c = i & (INTER_ - 1);
        sq[i] = qb[i] + bq[c];
        sk[i] = kbuf[i] + bk[c];
    }
    __syncthreads();
    {
        const int t = tid >> 4, u = tid & 15;
        float s = 0.f;
        if (t > u) {   // strict lower triangle, scaled by 1/gamma
            for (int i = 0; i < INTER_; ++i) s += sq[t * INTER_ + i] * sk[u * INTER_ + i];
            s *= (1.0f / 0.9f);
        }
        ss[tid] = s;
    }
    __syncthreads();
    for (int i = tid; i < TD_ * INTER_; i += 256) {
        const int t = i >> 8, c = i & (INTER_ - 1);
        float s = 0.f;
#pragma unroll
        for (int u = 0; u < TD_; ++u) s += ss[t * TD_ + u] * (vbuf[u * INTER_ + c] + bv[c]);
        ret[i] = s;
    }
}

// ---------------------------------------------------------------------------
// Kernel 5: eta (prelu(ret@Wret.T+bret) reshaped), skip, hp = cat @ rl2_w.T + rl2_b
// grid N, block 64
// ---------------------------------------------------------------------------
__global__ __launch_bounds__(64)
void hp_kernel(const float* __restrict__ ret, const float* __restrict__ Wret,
               const float* __restrict__ bret, const float* __restrict__ pretp,
               const float* __restrict__ x,
               const float* __restrict__ rl1_w0, const float* __restrict__ rl1_b0,
               const float* __restrict__ hp_prev,
               const float* __restrict__ rl1_w1, const float* __restrict__ rl1_b1,
               const float* __restrict__ rl2_w, const float* __restrict__ rl2_b,
               int layer, float* __restrict__ hp_out)
{
    __shared__ float sret[INTER_];
    __shared__ float sx[256];
    __shared__ float cat[80];
    const int n = blockIdx.x;
    const int o = threadIdx.x;
    const int t = n >> 7;
    for (int i = o; i < INTER_; i += 64) sret[i] = ret[t * INTER_ + i];
    if (layer == 0) {
        for (int i = o; i < 256; i += 64)
            sx[i] = x[((size_t)(i >> 6) * N_ + n) * F_ + (i & 63)];
    } else {
        sx[o] = hp_prev[(size_t)n * 64 + o];
    }
    __syncthreads();
    const float pret = pretp[0];
    if (o < 16) {
        const int oo = ((n & 127) << 4) | o;
        const float* w = Wret + (size_t)oo * INTER_;
        float s = bret[oo];
        for (int i = 0; i < INTER_; ++i) s += sret[i] * w[i];
        cat[o] = prelu(s, pret);
    }
    float s;
    if (layer == 0) {
        s = rl1_b0[o];
        const float* w = rl1_w0 + o * 256;
        for (int i = 0; i < 256; ++i) s += sx[i] * w[i];
    } else {
        s = rl1_b1[o];
        const float* w = rl1_w1 + o * 64;
#pragma unroll 8
        for (int i = 0; i < 64; ++i) s += sx[i] * w[i];
    }
    cat[16 + o] = s;
    __syncthreads();
    const float* w2 = rl2_w + o * 80;
    float hv = rl2_b[o];
#pragma unroll 8
    for (int i = 0; i < 80; ++i) hv += cat[i] * w2[i];
    hp_out[(size_t)n * 64 + o] = hv;
}

// ---------------------------------------------------------------------------
// Kernel 6: out = (prelu(hp@w1.T+b1) @ w2.T) + b2.  grid N, block 128.
// ---------------------------------------------------------------------------
__global__ __launch_bounds__(128)
void mlp_kernel(const float* __restrict__ hp,
                const float* __restrict__ w1, const float* __restrict__ b1,
                const float* __restrict__ w2, const float* __restrict__ b2,
                const float* __restrict__ pm, float* __restrict__ out)
{
    __shared__ float sh[64];
    __shared__ float sc[128];
    const int n = blockIdx.x;
    const int j = threadIdx.x;
    if (j < 64) sh[j] = hp[(size_t)n * 64 + j];
    __syncthreads();
    const float p = pm[0];
    float s = b1[j];
    const float* w = w1 + j * 64;
#pragma unroll 8
    for (int i = 0; i < 64; ++i) s += sh[i] * w[i];
    sc[j] = prelu(s, p);
    __syncthreads();
    if (j < 2) {
        float s2 = b2[j];
        const float* ww = w2 + j * 128;
#pragma unroll 8
        for (int i = 0; i < 128; ++i) s2 += sc[i] * ww[i];
        out[(size_t)n * 2 + j] = s2;
    }
}

// ---------------------------------------------------------------------------
extern "C" void kernel_launch(void* const* d_in, const int* in_sizes, int n_in,
                              void* d_out, int out_size, void* d_ws, size_t ws_size,
                              hipStream_t stream) {
    const float* x        = (const float*)d_in[0];
    const float* a        = (const float*)d_in[1];
    const float* T_param  = (const float*)d_in[2];
    const float* theta    = (const float*)d_in[3];
    const float* fc_w     = (const float*)d_in[4];
    const float* fc_b     = (const float*)d_in[5];
    const float* conv_w   = (const float*)d_in[6];
    const float* conv_b   = (const float*)d_in[7];
    const float* prelu0   = (const float*)d_in[8];
    const float* prelu1   = (const float*)d_in[9];
    const float* Wq       = (const float*)d_in[10];
    const float* bq       = (const float*)d_in[11];
    const float* Wk       = (const float*)d_in[12];
    const float* bk       = (const float*)d_in[13];
    const float* Wv       = (const float*)d_in[14];
    const float* bv       = (const float*)d_in[15];
    const float* Wret     = (const float*)d_in[16];
    const float* bret     = (const float*)d_in[17];
    const float* prelu_ret= (const float*)d_in[18];
    const float* rl1_w0   = (const float*)d_in[19];
    const float* rl1_b0   = (const float*)d_in[20];
    const float* rl1_w1   = (const float*)d_in[21];
    const float* rl1_b1   = (const float*)d_in[22];
    const float* rl2_w    = (const float*)d_in[23];
    const float* rl2_b    = (const float*)d_in[24];
    const float* mlp_w1   = (const float*)d_in[25];
    const float* mlp_b1   = (const float*)d_in[26];
    const float* mlp_w2   = (const float*)d_in[27];
    const float* mlp_b2   = (const float*)d_in[28];
    const float* prelu_mlp= (const float*)d_in[29];

    float* ws       = (float*)d_ws;
    float* featpart = ws;                                        // FSPLIT_*R*N*F = 4194304
    float* diffb    = featpart + (size_t)FSPLIT_ * R_ * N_ * F_;
    float* hbuf     = diffb + (size_t)R_ * N_ * F_;
    float* partb    = hbuf  + (size_t)R_ * N_ * F_;              // 3*KSPL_*4096 = 1572864
    float* qkvb     = partb + (size_t)3 * KSPL_ * TD_ * INTER_;  // 12288
    float* retb     = qkvb  + (size_t)3 * TD_ * INTER_;          // 4096
    float* hpA      = retb  + (size_t)TD_ * INTER_;              // 131072
    float* hpB      = hpA   + (size_t)N_ * 64;

    for (int l = 0; l < L_; ++l) {
        const float* hcur = (l == 0) ? x : hbuf;

        feat_wmma_kernel<<<dim3(N_ / 64, R_, FSPLIT_), 128, 0, stream>>>(
            hcur, a,
            T_param + (size_t)l * R_ * S_ * N_ * N_,
            theta + (size_t)l * R_ * S_,
            featpart);

        fc_conv_kernel<<<N_, 64, 0, stream>>>(
            featpart,
            fc_w + (size_t)l * R_ * F_ * F_, fc_b + (size_t)l * R_ * F_,
            conv_w + (size_t)l * R_ * R_, conv_b + (size_t)l * R_,
            prelu0 + l, prelu1 + l,
            diffb, hbuf, (l == 0) ? 1 : 0);

        qkv_wmma_kernel<<<dim3(KSPL_, INTER_ / 16, 3), 32, 0, stream>>>(
            diffb,
            Wq + (size_t)l * INTER_ * INDIM_,
            Wk + (size_t)l * INTER_ * INDIM_,
            Wv + (size_t)l * INTER_ * INDIM_,
            partb);

        qkv_reduce_kernel<<<dim3((TD_ * INTER_) / 256, 3), 256, 0, stream>>>(
            partb, qkvb);

        retention_kernel<<<1, 256, 0, stream>>>(
            qkvb, qkvb + TD_ * INTER_, qkvb + 2 * TD_ * INTER_,
            bq + (size_t)l * INTER_, bk + (size_t)l * INTER_, bv + (size_t)l * INTER_,
            retb);

        hp_kernel<<<N_, 64, 0, stream>>>(
            retb, Wret + (size_t)l * OUT_ * INTER_, bret + (size_t)l * OUT_,
            prelu_ret + l, x, rl1_w0, rl1_b0,
            hpA, rl1_w1, rl1_b1,
            rl2_w + (size_t)l * 64 * 80, rl2_b + (size_t)l * 64,
            l, (l == 0) ? hpA : hpB);
    }

    mlp_kernel<<<N_, 128, 0, stream>>>(
        hpB, mlp_w1, mlp_b1, mlp_w2, mlp_b2, prelu_mlp, (float*)d_out);
}
